// PPNet_wholeprotos_44813688767262
// MI455X (gfx1250) — compile-verified, matrix operand order checked by reading, not run on hardware
//
#include <hip/hip_runtime.h>
#include <hip/hip_bf16.h>

typedef __attribute__((ext_vector_type(16))) _Float16 v16h;
typedef __attribute__((ext_vector_type(8)))  _Float16 v8h;
typedef __attribute__((ext_vector_type(8)))  float    v8f;

#define NB 32
#define NC 128
#define NM 49
#define NP 512
#define REGC 0.1f
#define NIT 100
#define TSTR 136                     // padded LDS tile row stride (halves); 68-dword stride
#define LOGA (-3.8918202981106265f)  // log(1/49)

__global__ __launch_bounds__(128)
void ppnet_sinkhorn_kernel(const float* __restrict__ conv,
                           const float* __restrict__ prot,
                           const float* __restrict__ embS,
                           float* __restrict__ outMinD,
                           float* __restrict__ outOT)
{
    __shared__ _Float16 sA[64 * TSTR];   // feature tile  [64 x 128] f16, padded
    __shared__ _Float16 sB[64 * TSTR];   // prototype tile
    __shared__ float    sK[NM * NM];     // Kneg = -D/reg
    __shared__ float    sU[NM];
    __shared__ float    sV[NM];
    __shared__ float    sFn[64];
    __shared__ float    sPn[64];
    __shared__ float    sRed[4];

    const int tid  = threadIdx.x;
    const int lane = tid & 31;
    const int w    = tid >> 5;
    const int bp   = blockIdx.x;
    const int b    = bp >> 9;      // / 512
    const int p    = bp & 511;

    const float* cf = conv + (size_t)b * NC * NM;   // [C][49]
    const float* pf = prot + (size_t)p * NC * NM;   // [C][49]

    // ---- Phase 1: load + transpose (C,hw) -> (patch, C) as f16 into LDS ----
    for (int idx = tid; idx < NC * NM; idx += 128) {
        int c = idx / NM;
        int i = idx - c * NM;
        sA[i * TSTR + c] = (_Float16)cf[idx];   // coalesced global reads
        sB[i * TSTR + c] = (_Float16)pf[idx];
    }
    for (int idx = tid; idx < (64 - NM) * NC; idx += 128) {  // zero pad rows 49..63
        int rr = idx / NC;
        int c  = idx - rr * NC;
        sA[(NM + rr) * TSTR + c] = (_Float16)0.f;
        sB[(NM + rr) * TSTR + c] = (_Float16)0.f;
    }
    __syncthreads();

    // ---- Phase 2: squared norms (f32 accumulate over f16 tiles) ----
    if (tid < NM) {
        float fn = 0.f, pn = 0.f;
        for (int c = 0; c < NC; ++c) {
            float a  = (float)sA[tid * TSTR + c];  fn += a * a;
            float bb = (float)sB[tid * TSTR + c];  pn += bb * bb;
        }
        sFn[tid] = fn;  sPn[tid] = pn;
        sU[tid] = 0.f;  sV[tid] = 0.f;
    }
    __syncthreads();

    // ---- Phase 3: WMMA GEMM  acc = feat_tile * prot_tile^T  -> Kneg ----
    {
        const int mi   = w;                        // each wave owns one 16-row strip
        const int M    = mi * 16 + (lane & 15);
        const int koff = (lane & 16) ? 8 : 0;      // ISA 16-bit A/B layout: hi lanes hold K+8
        for (int ni = 0; ni < 4; ++ni) {
            const int N = ni * 16 + (lane & 15);
            v8f acc = {};
            #pragma unroll
            for (int kk = 0; kk < 4; ++kk) {
                const int kb = kk * 32 + koff;
                v8h a0 = *(const v8h*)(sA + M * TSTR + kb);        // K = kb..kb+7
                v8h a1 = *(const v8h*)(sA + M * TSTR + kb + 16);   // K = kb+16..kb+23
                v8h b0 = *(const v8h*)(sB + N * TSTR + kb);
                v8h b1 = *(const v8h*)(sB + N * TSTR + kb + 16);
                v16h af, bf;
                #pragma unroll
                for (int t = 0; t < 8; ++t) {
                    af[t] = a0[t]; af[t + 8] = a1[t];
                    bf[t] = b0[t]; bf[t + 8] = b1[t];
                }
                acc = __builtin_amdgcn_wmma_f32_16x16x32_f16(
                        false, af, false, bf, (short)0, acc, false, false);
            }
            const int jj    = ni * 16 + (lane & 15);
            const int ibase = mi * 16 + ((lane & 16) ? 8 : 0);
            #pragma unroll
            for (int rr = 0; rr < 8; ++rr) {
                const int ii = ibase + rr;
                if (ii < NM && jj < NM) {
                    float d2 = sFn[ii] + sPn[jj] - 2.0f * acc[rr];
                    float d  = sqrtf(fmaxf(d2, 0.f));
                    sK[ii * NM + jj] = -d * (1.0f / REGC);
                }
            }
        }
    }
    __syncthreads();

    // ---- Phase 4: log-domain Sinkhorn, 100 iterations, all on-chip ----
    for (int it = 0; it < NIT; ++it) {
        // u update: logsumexp over columns of each row
        for (int i = w; i < NM; i += 4) {
            float x0 = sK[i * NM + lane] + sV[lane];
            const bool has1 = (lane + 32) < NM;
            float x1 = has1 ? (sK[i * NM + lane + 32] + sV[lane + 32]) : -3.0e38f;
            float mx = fmaxf(x0, x1);
            #pragma unroll
            for (int off = 16; off; off >>= 1) mx = fmaxf(mx, __shfl_xor(mx, off, 32));
            float s = __expf(x0 - mx);
            if (has1) s += __expf(x1 - mx);
            #pragma unroll
            for (int off = 16; off; off >>= 1) s += __shfl_xor(s, off, 32);
            if (lane == 0) sU[i] += LOGA - (mx + __logf(s));
        }
        __syncthreads();
        // v update: logsumexp over rows of each column (odd LDS stride -> no conflicts)
        for (int j = w; j < NM; j += 4) {
            float x0 = sK[lane * NM + j] + sU[lane];
            const bool has1 = (lane + 32) < NM;
            float x1 = has1 ? (sK[(lane + 32) * NM + j] + sU[lane + 32]) : -3.0e38f;
            float mx = fmaxf(x0, x1);
            #pragma unroll
            for (int off = 16; off; off >>= 1) mx = fmaxf(mx, __shfl_xor(mx, off, 32));
            float s = __expf(x0 - mx);
            if (has1) s += __expf(x1 - mx);
            #pragma unroll
            for (int off = 16; off; off >>= 1) s += __shfl_xor(s, off, 32);
            if (lane == 0) sV[j] += LOGA - (mx + __logf(s));
        }
        __syncthreads();
    }

    // ---- Phase 5: OT = exp(Kneg+u+v), fused sum(OT*D) ----
    float part = 0.f;
    float* ot = outOT + (size_t)bp * (NM * NM);
    for (int idx = tid; idx < NM * NM; idx += 128) {
        int i = idx / NM;
        int j = idx - i * NM;
        float kn = sK[idx];
        float t  = __expf(kn + sU[i] + sV[j]);
        ot[idx]  = t;
        part += t * (-REGC * kn);     // D = -reg * Kneg
    }
    #pragma unroll
    for (int off = 16; off; off >>= 1) part += __shfl_xor(part, off, 32);
    if (lane == 0) sRed[w] = part;
    __syncthreads();
    if (tid == 0) {
        float tot = sRed[0] + sRed[1] + sRed[2] + sRed[3];
        outMinD[bp] = tot * embS[0];
    }
}

__global__ __launch_bounds__(128)
void ppnet_pred_kernel(const float* __restrict__ mind,
                       const float* __restrict__ rr,
                       const float* __restrict__ classes,
                       float* __restrict__ outPred)
{
    __shared__ float sN[4], sD[4];
    const int b = blockIdx.x;
    const int tid = threadIdx.x, lane = tid & 31, w = tid >> 5;
    const float rv = rr[0];
    const float inv_s2 = 1.0f / (2.0f * (rv / 3.0f) * (rv / 3.0f));
    float num = 0.f, den = 0.f;
    for (int p = tid; p < NP; p += 128) {
        float d  = mind[b * NP + p];
        float wt = (d > rv) ? 0.f : __expf(-(d * d) * inv_s2);
        num += classes[p] * wt;
        den += wt;
    }
    #pragma unroll
    for (int off = 16; off; off >>= 1) {
        num += __shfl_xor(num, off, 32);
        den += __shfl_xor(den, off, 32);
    }
    if (lane == 0) { sN[w] = num; sD[w] = den; }
    __syncthreads();
    if (tid == 0)
        outPred[b] = (sN[0] + sN[1] + sN[2] + sN[3]) / (sD[0] + sD[1] + sD[2] + sD[3]);
}

extern "C" void kernel_launch(void* const* d_in, const int* in_sizes, int n_in,
                              void* d_out, int out_size, void* d_ws, size_t ws_size,
                              hipStream_t stream) {
    (void)in_sizes; (void)n_in; (void)out_size; (void)d_ws; (void)ws_size;
    const float* conv = (const float*)d_in[0];   // [32,128,7,7]
    const float* prot = (const float*)d_in[1];   // [512,128,7,7]
    const float* embS = (const float*)d_in[2];   // [1]
    const float* r    = (const float*)d_in[3];   // [1]
    const float* cls  = (const float*)d_in[4];   // [512]

    float* out     = (float*)d_out;
    float* outMinD = out;                                           // [32,512]
    float* outOT   = out + (size_t)NB * NP;                         // [32,512,49,49]
    float* outPred = out + (size_t)NB * NP + (size_t)NB * NP * NM * NM;  // [32]

    ppnet_sinkhorn_kernel<<<NB * NP, 128, 0, stream>>>(conv, prot, embS, outMinD, outOT);
    ppnet_pred_kernel<<<NB, 128, 0, stream>>>(outMinD, r, cls, outPred);
}
